// Llama4MoE_5093831213309
// MI455X (gfx1250) — compile-verified, matrix operand order checked by reading, not run on hardware
//
#include <hip/hip_runtime.h>
#include <cstdint>

#define TT 4096   // B*S tokens
#define HH 2048   // hidden
#define II 2048   // intermediate
#define EE 8      // routed experts
#define MCHUNKS (TT / 128)   // 32

typedef __attribute__((ext_vector_type(16))) __bf16 v16bf;
typedef __attribute__((ext_vector_type(8)))  float  v8f;

// ---------------- gfx1250 async-to-LDS staging ----------------

#if defined(__HIP_DEVICE_COMPILE__) && \
    __has_builtin(__builtin_amdgcn_global_load_async_to_lds_b128) && \
    __has_builtin(__builtin_amdgcn_s_wait_asynccnt)
#define USE_ASYNC 1
#else
#define USE_ASYNC 0
#endif

#if USE_ASYNC
typedef int v4i_vs __attribute__((vector_size(16)));
typedef v4i_vs __attribute__((address_space(1))) g_v4i;   // global
typedef v4i_vs __attribute__((address_space(3))) l_v4i;   // LDS
#endif

__device__ __forceinline__ void stage16(const void* g, void* l) {
#if USE_ASYNC
  __builtin_amdgcn_global_load_async_to_lds_b128(
      (g_v4i*)(uintptr_t)g,
      (l_v4i*)(uint32_t)(uintptr_t)l,
      0, 0);
#else
  *(uint4*)l = *(const uint4*)g;
#endif
}

__device__ __forceinline__ void stage_wait() {
#if USE_ASYNC
  __builtin_amdgcn_s_wait_asynccnt(0);
#endif
}

// ---------------- helpers ----------------

__device__ __forceinline__ unsigned short f2bf(float f) {
  uint32_t u = __builtin_bit_cast(uint32_t, f);
  uint32_t r = u + 0x7FFFu + ((u >> 16) & 1u);   // round-to-nearest-even
  return (unsigned short)(r >> 16);
}

// Read one lane's half of a 16x32 bf16 WMMA operand from an LDS row (32 elems, 64B).
// hi=0: elems {k0..k0+7, k0+16..k0+23}; hi=1: {k0+8..+15, k0+24..+31}
__device__ __forceinline__ v16bf lds_frag(const unsigned short* row, int hi) {
  union { v16bf v; uint4 q[2]; } f;
  const uint4* p = reinterpret_cast<const uint4*>(row + (hi << 3));
  f.q[0] = p[0];
  f.q[1] = p[2];
  return f.v;
}

__device__ __forceinline__ v8f zero_acc() {
  v8f c;
#pragma unroll
  for (int i = 0; i < 8; ++i) c[i] = 0.0f;
  return c;
}

__device__ __forceinline__ float sigmoid_fast(float x) { return 1.0f / (1.0f + __expf(-x)); }
__device__ __forceinline__ float silu_fast(float x)    { return x / (1.0f + __expf(-x)); }

#define WMMA_BF16(a, b, c) \
  __builtin_amdgcn_wmma_f32_16x16x32_bf16(false, (a), false, (b), (short)0, (c), false, false)

// ---------------- small kernels ----------------

__global__ void moe_zero_kernel(int* counts, int* cursors) {
  int i = threadIdx.x;
  if (i < EE) { counts[i] = 0; cursors[i] = 0; }
}

__global__ void convert_f32_bf16_v4(const float4* __restrict__ src,
                                    ushort4* __restrict__ dst, long n4) {
  long i = (long)blockIdx.x * blockDim.x + threadIdx.x;
  long stride = (long)gridDim.x * blockDim.x;
  for (; i < n4; i += stride) {
    float4 v = src[i];
    ushort4 o;
    o.x = f2bf(v.x); o.y = f2bf(v.y); o.z = f2bf(v.z); o.w = f2bf(v.w);
    dst[i] = o;
  }
}

__global__ __launch_bounds__(256) void moe_router_kernel(
    const float* __restrict__ x, const float* __restrict__ gw,
    float* __restrict__ logits_out,
    unsigned short* __restrict__ xs_bf, unsigned short* __restrict__ x_bf,
    int* __restrict__ top_id, int* __restrict__ counts) {
  int t = blockIdx.x * (blockDim.x >> 5) + (threadIdx.x >> 5);
  int lane = threadIdx.x & 31;
  if (t >= TT) return;
  const float* xr = x + (size_t)t * HH;
  float acc[EE];
#pragma unroll
  for (int e = 0; e < EE; ++e) acc[e] = 0.0f;
  for (int h = lane; h < HH; h += 32) {
    float xv = xr[h];
#pragma unroll
    for (int e = 0; e < EE; ++e) acc[e] += xv * gw[e * HH + h];
  }
#pragma unroll
  for (int e = 0; e < EE; ++e)
#pragma unroll
    for (int off = 16; off; off >>= 1) acc[e] += __shfl_xor(acc[e], off, 32);
  float best = acc[0]; int bid = 0;
#pragma unroll
  for (int e = 1; e < EE; ++e)
    if (acc[e] > best) { best = acc[e]; bid = e; }   // first-max wins (argmax semantics)
  float scale = sigmoid_fast(best);
  if (lane == 0) {
#pragma unroll
    for (int e = 0; e < EE; ++e) logits_out[(size_t)t * EE + e] = acc[e];
    top_id[t] = bid;
    atomicAdd(&counts[bid], 1);
  }
  for (int h = lane; h < HH; h += 32) {
    float xv = xr[h];
    xs_bf[(size_t)t * HH + h] = f2bf(xv * scale);
    x_bf [(size_t)t * HH + h] = f2bf(xv);
  }
}

__global__ void moe_scan_kernel(const int* __restrict__ counts, int* __restrict__ offs) {
  if (threadIdx.x == 0) {
    int r = 0;
    for (int e = 0; e < EE; ++e) { offs[e] = r; r += counts[e]; }
    offs[EE] = r;
  }
}

__global__ void moe_scatter_kernel(const int* __restrict__ top_id, const int* __restrict__ offs,
                                   int* __restrict__ cursors, int* __restrict__ order) {
  int t = blockIdx.x * blockDim.x + threadIdx.x;
  if (t < TT) {
    int e = top_id[t];
    int pos = offs[e] + atomicAdd(&cursors[e], 1);
    order[pos] = t;
  }
}

// ---------------- WMMA GEMM kernels (LDS double-buffered) ----------------
// Block = 8 waves -> 128 rows x 128 cols. Per K-step of 32: A tile + B tile(s)
// staged into LDS (async when available), all 8 waves share the B tiles.

// routed gate+up fused, SiLU(g)*u -> bf16 h (expert-sorted rows)
__global__ __launch_bounds__(256) void moe_mlp1_routed(
    const unsigned short* __restrict__ xs,     // [T,H] bf16, token order
    const unsigned short* __restrict__ wg,     // [E,I,H] bf16
    const unsigned short* __restrict__ wu,     // [E,I,H] bf16
    const int* __restrict__ order, const int* __restrict__ offs,
    unsigned short* __restrict__ hout) {       // [T,I] bf16, sorted order
  __shared__ __align__(16) unsigned short lds[2 * 3 * 4096];   // 48 KB
  int e = blockIdx.x >> 5;            // / MCHUNKS
  int chunk = blockIdx.x & 31;
  int gbeg = offs[e], gend = offs[e + 1];
  int m0 = gbeg + chunk * 128;
  if (m0 >= gend) return;             // block-uniform, before any barrier
  int wave = threadIdx.x >> 5, lane = threadIdx.x & 31;
  int l15 = lane & 15, hi = lane >> 4;
  int n0 = blockIdx.y * 128;
  int maxrow = gend - 1;
  // per-lane staging bases: 48 slots of 512B (A | Bgate | Bup), 6 per wave
  const char* gb[6]; int lo[6];
#pragma unroll
  for (int s = 0; s < 6; ++s) {
    int sid = wave * 6 + s;
    int byt = sid * 512 + lane * 16;
    int tile = byt >> 13;
    int o = byt & 8191;
    int row = o >> 6, colb = o & 63;
    if (tile == 0) {
      int ar = m0 + row; if (ar > maxrow) ar = maxrow;   // clamp ragged tail
      gb[s] = (const char*)(xs + (size_t)order[ar] * HH) + colb;
    } else {
      const unsigned short* w = (tile == 1) ? wg : wu;
      gb[s] = (const char*)(w + ((size_t)e * II + n0 + row) * HH) + colb;
    }
    lo[s] = tile * 8192 + row * 64 + colb;
  }
  char* lb = (char*)lds;
  v8f g[8], u[8];
#pragma unroll
  for (int j = 0; j < 8; ++j) { g[j] = zero_acc(); u[j] = zero_acc(); }
#pragma unroll
  for (int s = 0; s < 6; ++s) stage16(gb[s], lb + lo[s]);
  const int arow = (wave * 16 + l15) * 32;
  for (int kt = 0; kt < HH / 32; ++kt) {
    stage_wait();
    __syncthreads();
    int par = (kt & 1) * 12288;                      // element offset
    if (kt + 1 < HH / 32) {
      int npb = ((kt + 1) & 1) * 24576;              // byte offset
#pragma unroll
      for (int s = 0; s < 6; ++s)
        stage16(gb[s] + (size_t)(kt + 1) * 64, lb + npb + lo[s]);
    }
    v16bf a = lds_frag(lds + par + arow, hi);
    v16bf bf[8];
#pragma unroll
    for (int j = 0; j < 8; ++j) bf[j] = lds_frag(lds + par + 4096 + (j * 16 + l15) * 32, hi);
#pragma unroll
    for (int j = 0; j < 8; ++j) g[j] = WMMA_BF16(a, bf[j], g[j]);
#pragma unroll
    for (int j = 0; j < 8; ++j) bf[j] = lds_frag(lds + par + 8192 + (j * 16 + l15) * 32, hi);
#pragma unroll
    for (int j = 0; j < 8; ++j) u[j] = WMMA_BF16(a, bf[j], u[j]);
    __syncthreads();
  }
  int mb = m0 + wave * 16;
#pragma unroll
  for (int j = 0; j < 8; ++j)
#pragma unroll
    for (int i = 0; i < 8; ++i) {
      int m = mb + i + hi * 8;
      if (m < gend) {
        float hv = silu_fast(g[j][i]) * u[j][i];
        hout[(size_t)m * II + n0 + j * 16 + l15] = f2bf(hv);
      }
    }
}

// routed down: out[token] = h @ Wd_e^T (plain store; top-1 covers each token once)
__global__ __launch_bounds__(256) void moe_mlp2_routed(
    const unsigned short* __restrict__ hin,    // [T,I] bf16, sorted order
    const unsigned short* __restrict__ wd,     // [E,H,I] bf16
    const int* __restrict__ order, const int* __restrict__ offs,
    float* __restrict__ out) {                 // [T,H] fp32, token order
  __shared__ __align__(16) unsigned short lds[2 * 2 * 4096];   // 32 KB
  int e = blockIdx.x >> 5;
  int chunk = blockIdx.x & 31;
  int gbeg = offs[e], gend = offs[e + 1];
  int m0 = gbeg + chunk * 128;
  if (m0 >= gend) return;
  int wave = threadIdx.x >> 5, lane = threadIdx.x & 31;
  int l15 = lane & 15, hi = lane >> 4;
  int n0 = blockIdx.y * 128;
  const char* gb[4]; int lo[4];
#pragma unroll
  for (int s = 0; s < 4; ++s) {
    int sid = wave * 4 + s;
    int byt = sid * 512 + lane * 16;
    int tile = byt >> 13;
    int o = byt & 8191;
    int row = o >> 6, colb = o & 63;
    if (tile == 0) {
      int ar = m0 + row; if (ar > TT - 1) ar = TT - 1;
      gb[s] = (const char*)(hin + (size_t)ar * II) + colb;
    } else {
      gb[s] = (const char*)(wd + ((size_t)e * HH + n0 + row) * II) + colb;
    }
    lo[s] = tile * 8192 + row * 64 + colb;
  }
  char* lb = (char*)lds;
  v8f acc[8];
#pragma unroll
  for (int j = 0; j < 8; ++j) acc[j] = zero_acc();
#pragma unroll
  for (int s = 0; s < 4; ++s) stage16(gb[s], lb + lo[s]);
  const int arow = (wave * 16 + l15) * 32;
  for (int kt = 0; kt < II / 32; ++kt) {
    stage_wait();
    __syncthreads();
    int par = (kt & 1) * 8192;                       // element offset
    if (kt + 1 < II / 32) {
      int npb = ((kt + 1) & 1) * 16384;              // byte offset
#pragma unroll
      for (int s = 0; s < 4; ++s)
        stage16(gb[s] + (size_t)(kt + 1) * 64, lb + npb + lo[s]);
    }
    v16bf a = lds_frag(lds + par + arow, hi);
    v16bf bf[8];
#pragma unroll
    for (int j = 0; j < 8; ++j) bf[j] = lds_frag(lds + par + 4096 + (j * 16 + l15) * 32, hi);
#pragma unroll
    for (int j = 0; j < 8; ++j) acc[j] = WMMA_BF16(a, bf[j], acc[j]);
    __syncthreads();
  }
  int mb = m0 + wave * 16;
#pragma unroll
  for (int j = 0; j < 8; ++j)
#pragma unroll
    for (int i = 0; i < 8; ++i) {
      int m = mb + i + hi * 8;
      if (m < gend) {
        int tok = order[m];
        out[(size_t)tok * HH + n0 + j * 16 + l15] = acc[j][i];
      }
    }
}

// shared gate+up fused
__global__ __launch_bounds__(256) void moe_mlp1_shared(
    const unsigned short* __restrict__ xb,     // [T,H] bf16
    const unsigned short* __restrict__ wg,     // [I,H] bf16
    const unsigned short* __restrict__ wu,     // [I,H] bf16
    unsigned short* __restrict__ hout) {       // [T,I] bf16
  __shared__ __align__(16) unsigned short lds[2 * 3 * 4096];
  int m0 = blockIdx.x * 128;
  int wave = threadIdx.x >> 5, lane = threadIdx.x & 31;
  int l15 = lane & 15, hi = lane >> 4;
  int n0 = blockIdx.y * 128;
  const char* gb[6]; int lo[6];
#pragma unroll
  for (int s = 0; s < 6; ++s) {
    int sid = wave * 6 + s;
    int byt = sid * 512 + lane * 16;
    int tile = byt >> 13;
    int o = byt & 8191;
    int row = o >> 6, colb = o & 63;
    if (tile == 0) {
      gb[s] = (const char*)(xb + (size_t)(m0 + row) * HH) + colb;
    } else {
      const unsigned short* w = (tile == 1) ? wg : wu;
      gb[s] = (const char*)(w + (size_t)(n0 + row) * HH) + colb;
    }
    lo[s] = tile * 8192 + row * 64 + colb;
  }
  char* lb = (char*)lds;
  v8f g[8], u[8];
#pragma unroll
  for (int j = 0; j < 8; ++j) { g[j] = zero_acc(); u[j] = zero_acc(); }
#pragma unroll
  for (int s = 0; s < 6; ++s) stage16(gb[s], lb + lo[s]);
  const int arow = (wave * 16 + l15) * 32;
  for (int kt = 0; kt < HH / 32; ++kt) {
    stage_wait();
    __syncthreads();
    int par = (kt & 1) * 12288;
    if (kt + 1 < HH / 32) {
      int npb = ((kt + 1) & 1) * 24576;
#pragma unroll
      for (int s = 0; s < 6; ++s)
        stage16(gb[s] + (size_t)(kt + 1) * 64, lb + npb + lo[s]);
    }
    v16bf a = lds_frag(lds + par + arow, hi);
    v16bf bf[8];
#pragma unroll
    for (int j = 0; j < 8; ++j) bf[j] = lds_frag(lds + par + 4096 + (j * 16 + l15) * 32, hi);
#pragma unroll
    for (int j = 0; j < 8; ++j) g[j] = WMMA_BF16(a, bf[j], g[j]);
#pragma unroll
    for (int j = 0; j < 8; ++j) bf[j] = lds_frag(lds + par + 8192 + (j * 16 + l15) * 32, hi);
#pragma unroll
    for (int j = 0; j < 8; ++j) u[j] = WMMA_BF16(a, bf[j], u[j]);
    __syncthreads();
  }
  int mb = m0 + wave * 16;
#pragma unroll
  for (int j = 0; j < 8; ++j)
#pragma unroll
    for (int i = 0; i < 8; ++i) {
      int m = mb + i + hi * 8;
      float hv = silu_fast(g[j][i]) * u[j][i];
      hout[(size_t)m * II + n0 + j * 16 + l15] = f2bf(hv);
    }
}

// shared down: out += h_shared @ Wd^T
__global__ __launch_bounds__(256) void moe_mlp2_shared(
    const unsigned short* __restrict__ hin,    // [T,I] bf16
    const unsigned short* __restrict__ wd,     // [H,I] bf16
    float* __restrict__ out) {                 // [T,H] fp32 accumulate
  __shared__ __align__(16) unsigned short lds[2 * 2 * 4096];
  int m0 = blockIdx.x * 128;
  int wave = threadIdx.x >> 5, lane = threadIdx.x & 31;
  int l15 = lane & 15, hi = lane >> 4;
  int n0 = blockIdx.y * 128;
  const char* gb[4]; int lo[4];
#pragma unroll
  for (int s = 0; s < 4; ++s) {
    int sid = wave * 4 + s;
    int byt = sid * 512 + lane * 16;
    int tile = byt >> 13;
    int o = byt & 8191;
    int row = o >> 6, colb = o & 63;
    if (tile == 0) gb[s] = (const char*)(hin + (size_t)(m0 + row) * II) + colb;
    else           gb[s] = (const char*)(wd + (size_t)(n0 + row) * II) + colb;
    lo[s] = tile * 8192 + row * 64 + colb;
  }
  char* lb = (char*)lds;
  v8f acc[8];
#pragma unroll
  for (int j = 0; j < 8; ++j) acc[j] = zero_acc();
#pragma unroll
  for (int s = 0; s < 4; ++s) stage16(gb[s], lb + lo[s]);
  const int arow = (wave * 16 + l15) * 32;
  for (int kt = 0; kt < II / 32; ++kt) {
    stage_wait();
    __syncthreads();
    int par = (kt & 1) * 8192;
    if (kt + 1 < II / 32) {
      int npb = ((kt + 1) & 1) * 16384;
#pragma unroll
      for (int s = 0; s < 4; ++s)
        stage16(gb[s] + (size_t)(kt + 1) * 64, lb + npb + lo[s]);
    }
    v16bf a = lds_frag(lds + par + arow, hi);
    v16bf bf[8];
#pragma unroll
    for (int j = 0; j < 8; ++j) bf[j] = lds_frag(lds + par + 4096 + (j * 16 + l15) * 32, hi);
#pragma unroll
    for (int j = 0; j < 8; ++j) acc[j] = WMMA_BF16(a, bf[j], acc[j]);
    __syncthreads();
  }
  int mb = m0 + wave * 16;
#pragma unroll
  for (int j = 0; j < 8; ++j)
#pragma unroll
    for (int i = 0; i < 8; ++i) {
      int m = mb + i + hi * 8;
      size_t idx = (size_t)m * HH + n0 + j * 16 + l15;
      out[idx] += acc[j][i];
    }
}

// ---------------- launch ----------------

extern "C" void kernel_launch(void* const* d_in, const int* in_sizes, int n_in,
                              void* d_out, int out_size, void* d_ws, size_t ws_size,
                              hipStream_t stream) {
  (void)in_sizes; (void)n_in; (void)out_size; (void)ws_size;
  const float* x   = (const float*)d_in[0];
  const float* gw  = (const float*)d_in[1];
  const float* sgw = (const float*)d_in[2];
  const float* suw = (const float*)d_in[3];
  const float* sdw = (const float*)d_in[4];
  const float* rgw = (const float*)d_in[5];
  const float* ruw = (const float*)d_in[6];
  const float* rdw = (const float*)d_in[7];

  float* out    = (float*)d_out;                       // [T,H]
  float* logits = (float*)d_out + (size_t)TT * HH;     // [T,E]

  char* ws = (char*)d_ws;
  size_t off = 0;
  auto alloc = [&](size_t bytes) -> void* {
    void* p = ws + off;
    off += (bytes + 255) & ~(size_t)255;
    return p;
  };

  unsigned short* xs_bf  = (unsigned short*)alloc((size_t)TT * HH * 2);
  unsigned short* x_bf   = (unsigned short*)alloc((size_t)TT * HH * 2);
  unsigned short* h_rt   = (unsigned short*)alloc((size_t)TT * II * 2);
  unsigned short* h_sh   = (unsigned short*)alloc((size_t)TT * II * 2);
  unsigned short* sgw_bf = (unsigned short*)alloc((size_t)II * HH * 2);
  unsigned short* suw_bf = (unsigned short*)alloc((size_t)II * HH * 2);
  unsigned short* sdw_bf = (unsigned short*)alloc((size_t)HH * II * 2);
  unsigned short* rgw_bf = (unsigned short*)alloc((size_t)EE * II * HH * 2);
  unsigned short* ruw_bf = (unsigned short*)alloc((size_t)EE * II * HH * 2);
  unsigned short* rdw_bf = (unsigned short*)alloc((size_t)EE * HH * II * 2);
  int* top_id  = (int*)alloc((size_t)TT * 4);
  int* order   = (int*)alloc((size_t)TT * 4);
  int* counts  = (int*)alloc((size_t)EE * 4);
  int* cursors = (int*)alloc((size_t)EE * 4);
  int* goffs   = (int*)alloc((size_t)(EE + 1) * 4);

  moe_zero_kernel<<<1, 32, 0, stream>>>(counts, cursors);

  const long nS4 = (long)II * HH / 4;
  const long nR4 = (long)EE * II * HH / 4;
  convert_f32_bf16_v4<<<4096, 256, 0, stream>>>((const float4*)sgw, (ushort4*)sgw_bf, nS4);
  convert_f32_bf16_v4<<<4096, 256, 0, stream>>>((const float4*)suw, (ushort4*)suw_bf, nS4);
  convert_f32_bf16_v4<<<4096, 256, 0, stream>>>((const float4*)sdw, (ushort4*)sdw_bf, nS4);
  convert_f32_bf16_v4<<<8192, 256, 0, stream>>>((const float4*)rgw, (ushort4*)rgw_bf, nR4);
  convert_f32_bf16_v4<<<8192, 256, 0, stream>>>((const float4*)ruw, (ushort4*)ruw_bf, nR4);
  convert_f32_bf16_v4<<<8192, 256, 0, stream>>>((const float4*)rdw, (ushort4*)rdw_bf, nR4);

  moe_router_kernel<<<TT / 8, 256, 0, stream>>>(x, gw, logits, xs_bf, x_bf, top_id, counts);
  moe_scan_kernel<<<1, 32, 0, stream>>>(counts, goffs);
  moe_scatter_kernel<<<TT / 256, 256, 0, stream>>>(top_id, goffs, cursors, order);

  moe_mlp1_routed<<<dim3(EE * MCHUNKS, II / 128), 256, 0, stream>>>(xs_bf, rgw_bf, ruw_bf, order, goffs, h_rt);
  moe_mlp2_routed<<<dim3(EE * MCHUNKS, HH / 128), 256, 0, stream>>>(h_rt, rdw_bf, order, goffs, out);

  moe_mlp1_shared<<<dim3(MCHUNKS, II / 128), 256, 0, stream>>>(x_bf, sgw_bf, suw_bf, h_sh);
  moe_mlp2_shared<<<dim3(MCHUNKS, HH / 128), 256, 0, stream>>>(h_sh, sdw_bf, out);
}
